// DictMoE_80418967650852
// MI455X (gfx1250) — matrix operand order repaired and checked by reading, not compiled, not run on hardware
//
#include <hip/hip_runtime.h>
#include <math.h>

// ---------------------------------------------------------------------------
// DictMoE for MI455X (gfx1250, wave32, WMMA).
//   1) convert x, Wg1, Wg2 to bf16
//   2) router: two bf16 WMMA GEMMs with ReLU
//   3) per-token logits (E=8) + softmax + mean via wave reduction + atomics
//   4) coeff = gate + ohe ; copy gate to d_out tail
//   5) pre-merge per-sample weights: W1b/W2b -> bf16 (fits L2), b1b/b2b fp32
//   6) batched WMMA GEMM (exact GELU) -> hmid bf16
//   7) batched WMMA GEMM -> fp32 out
// GEMM: 128x128 block tile, 8 waves x (64x32), double-buffered LDS,
// A tile via async global->LDS copy (ASYNCcnt path).
// ---------------------------------------------------------------------------

typedef __attribute__((ext_vector_type(16))) __bf16 v16bf;
typedef __attribute__((ext_vector_type(8)))  float  v8f;

#define TM 128
#define TN 128
#define TK 32
#define LP 40   // LDS row pitch in bf16 elements (32 + 8 pad)

#if defined(__has_builtin)
#if __has_builtin(__builtin_amdgcn_global_load_async_to_lds_b128) && \
    __has_builtin(__builtin_amdgcn_s_wait_asynccnt)
#define USE_ASYNC_LDS 1
#endif
#endif
#ifndef USE_ASYNC_LDS
#define USE_ASYNC_LDS 0
#endif

typedef int vi4 __attribute__((vector_size(16)));         // clang native int4
typedef __attribute__((address_space(1))) vi4* g4_ptr;    // global vi4*
typedef __attribute__((address_space(3))) vi4* l4_ptr;    // LDS vi4*

__device__ __forceinline__ unsigned short f2bf(float f) {
  unsigned int u = __builtin_bit_cast(unsigned int, f);
  u += 0x7FFFu + ((u >> 16) & 1u);            // round-to-nearest-even
  return (unsigned short)(u >> 16);
}
__device__ __forceinline__ float bf2f(unsigned short h) {
  unsigned int u = ((unsigned int)h) << 16;
  return __builtin_bit_cast(float, u);
}

struct U32x8 { uint4 lo, hi; };
__device__ __forceinline__ v16bf mkfrag(uint4 lo, uint4 hi) {
  U32x8 t; t.lo = lo; t.hi = hi;
  return __builtin_bit_cast(v16bf, t);
}

// ---------------------------------------------------------------------------
// Generic bf16 GEMM: C[b] = act(A[b] (MxK) * B[b] (KxN) + bias[b])
// ACT: 0 = none, 1 = relu, 2 = exact gelu.  OUTF32: 1 = fp32 out, 0 = bf16 out
// ---------------------------------------------------------------------------
template <int ACT, int OUTF32>
__global__ __launch_bounds__(256) void gemm_bf16_wmma(
    const unsigned short* __restrict__ Ag,
    const unsigned short* __restrict__ Bg,
    const float* __restrict__ biasg,
    unsigned short* __restrict__ outB,
    float* __restrict__ outF,
    long long sA, long long sB, long long sBias, long long sC,
    int M, int N, int K)
{
  __shared__ unsigned short lA[2][TM * LP];
  __shared__ unsigned short lB[2][TN * LP];

  const int tid  = threadIdx.x;
  const int lane = tid & 31;
  const int wv   = tid >> 5;
  const int bz   = blockIdx.z;

  const unsigned short* A = Ag + (long long)bz * sA;
  const unsigned short* B = Bg + (long long)bz * sB;

  const int m0 = blockIdx.y * TM;
  const int n0 = blockIdx.x * TN;
  const int wm = (wv >> 2) * 64;   // wave row offset within tile
  const int wn = (wv & 3) * 32;    // wave col offset within tile

  v8f zero = {};
  v8f acc[4][2];
#pragma unroll
  for (int i = 0; i < 4; ++i)
#pragma unroll
    for (int j = 0; j < 2; ++j) acc[i][j] = zero;

  // A staging: each thread copies 16 bf16 of one row-segment
  const int arow = tid >> 1;          // 0..127
  const int aseg = (tid & 1) << 4;    // 0 or 16
  // B staging: each thread handles k-pair (kp,kp+1) x 8 columns
  const int kp   = (tid >> 4) << 1;   // 0,2,..,30
  const int bn8  = (tid & 15) << 3;   // 0..120
  const int lh   = lane >> 4;         // K-half selector
  const int ll   = lane & 15;

  auto loadTile = [&](int k0, int buf) {
    // ---- A tile: row-major [row][k] copy (no transpose) ----
    const unsigned short* ap = A + (long long)(m0 + arow) * K + (k0 + aseg);
    unsigned short* lap = &lA[buf][arow * LP + aseg];
#if USE_ASYNC_LDS
    __builtin_amdgcn_global_load_async_to_lds_b128((g4_ptr)ap, (l4_ptr)lap, 0, 0);
    __builtin_amdgcn_global_load_async_to_lds_b128((g4_ptr)(ap + 8), (l4_ptr)(lap + 8), 0, 0);
#else
    uint4 a0 = *(const uint4*)ap;
    uint4 a1 = *(const uint4*)(ap + 8);
    *(uint4*)lap       = a0;
    *(uint4*)(lap + 8) = a1;
#endif
    // ---- B tile: transpose to [n][k]; store packed (k,k+1) pairs as b32 ----
    const unsigned short* bp0 = B + (long long)(k0 + kp) * N + (n0 + bn8);
    uint4 r0 = *(const uint4*)bp0;
    uint4 r1 = *(const uint4*)(bp0 + N);
    const unsigned short* s0 = (const unsigned short*)&r0;
    const unsigned short* s1 = (const unsigned short*)&r1;
#pragma unroll
    for (int j = 0; j < 8; ++j) {
      unsigned int pk = (unsigned int)s0[j] | ((unsigned int)s1[j] << 16);
      *(unsigned int*)&lB[buf][(bn8 + j) * LP + kp] = pk;
    }
  };

  loadTile(0, 0);
#if USE_ASYNC_LDS
  __builtin_amdgcn_s_wait_asynccnt(0);
#endif
  __syncthreads();

  int cur = 0;
  for (int k0 = 0; k0 < K; k0 += TK) {
    if (k0 + TK < K) {
      __builtin_prefetch(A + (long long)(m0 + arow) * K + (k0 + 2 * TK + aseg), 0, 0);
      loadTile(k0 + TK, cur ^ 1);
    }

    // A frag: lane = row(M); lanes<16 hold K {0..7,16..23}, lanes>=16 {8..15,24..31}
    v16bf af[4];
#pragma unroll
    for (int mi = 0; mi < 4; ++mi) {
      const unsigned short* p = &lA[cur][(wm + mi * 16 + ll) * LP + lh * 8];
      af[mi] = mkfrag(*(const uint4*)p, *(const uint4*)(p + 16));
    }
    // B frag: lane = col(N); lanes<16 hold K {0..7,16..23}, lanes>=16 {8..15,24..31}
    v16bf bfr[2];
#pragma unroll
    for (int ni = 0; ni < 2; ++ni) {
      const unsigned short* p = &lB[cur][(wn + ni * 16 + ll) * LP + lh * 8];
      bfr[ni] = mkfrag(*(const uint4*)p, *(const uint4*)(p + 16));
    }

#pragma unroll
    for (int mi = 0; mi < 4; ++mi)
#pragma unroll
      for (int ni = 0; ni < 2; ++ni)
        acc[mi][ni] = __builtin_amdgcn_wmma_f32_16x16x32_bf16(
            false, af[mi], false, bfr[ni], (short)0, acc[mi][ni], false, false);

#if USE_ASYNC_LDS
    __builtin_amdgcn_s_wait_asynccnt(0);
#endif
    __syncthreads();
    cur ^= 1;
  }

  // epilogue: D frag lane = col, VGPR i -> row i + 8*(lane>=16)
  const float* bias = biasg ? (biasg + (long long)bz * sBias) : (const float*)0;
#pragma unroll
  for (int ni = 0; ni < 2; ++ni) {
    const int col = n0 + wn + ni * 16 + ll;
    const float bv = bias ? bias[col] : 0.f;
#pragma unroll
    for (int mi = 0; mi < 4; ++mi) {
#pragma unroll
      for (int i = 0; i < 8; ++i) {
        const int row = m0 + wm + mi * 16 + lh * 8 + i;
        float v = acc[mi][ni][i] + bv;
        if (ACT == 1) v = v > 0.f ? v : 0.f;
        if (ACT == 2) v = 0.5f * v * (1.f + erff(v * 0.70710678118654752f));
        const long long idx = (long long)bz * sC + (long long)row * N + col;
        if (OUTF32) outF[idx] = v;
        else        outB[idx] = f2bf(v);
      }
    }
  }
}

// ---------------------------------------------------------------------------
__global__ __launch_bounds__(256) void cvt_f32_to_bf16(
    const float* __restrict__ in, unsigned short* __restrict__ out, long long n)
{
  long long i = ((long long)blockIdx.x * 256 + threadIdx.x) * 4;
  if (i + 3 < n) {
    float4 v = *(const float4*)(in + i);
    unsigned long long p = (unsigned long long)f2bf(v.x)
        | ((unsigned long long)f2bf(v.y) << 16)
        | ((unsigned long long)f2bf(v.z) << 32)
        | ((unsigned long long)f2bf(v.w) << 48);
    *(unsigned long long*)(out + i) = p;
  }
}

// out_bf[b][i] = bf16( Wbase[i] + sum_e coeff[b,e] * T[e][i] )
__global__ __launch_bounds__(256) void merge_w_bf16(
    const float* __restrict__ Wb, const float* __restrict__ T,
    const float* __restrict__ coeff, unsigned short* __restrict__ out,
    long long n)
{
  const int b = blockIdx.y;
  long long i = ((long long)blockIdx.x * 256 + threadIdx.x) * 4;
  float c[8];
#pragma unroll
  for (int e = 0; e < 8; ++e) c[e] = coeff[b * 8 + e];
  float4 v = *(const float4*)(Wb + i);
#pragma unroll
  for (int e = 0; e < 8; ++e) {
    float4 t = *(const float4*)(T + (long long)e * n + i);
    v.x += c[e] * t.x; v.y += c[e] * t.y; v.z += c[e] * t.z; v.w += c[e] * t.w;
  }
  unsigned long long p = (unsigned long long)f2bf(v.x)
      | ((unsigned long long)f2bf(v.y) << 16)
      | ((unsigned long long)f2bf(v.z) << 32)
      | ((unsigned long long)f2bf(v.w) << 48);
  *(unsigned long long*)(out + (long long)b * n + i) = p;
}

__global__ __launch_bounds__(256) void merge_b_f32(
    const float* __restrict__ Wb, const float* __restrict__ T,
    const float* __restrict__ coeff, float* __restrict__ out, long long n)
{
  const int b = blockIdx.y;
  long long i = ((long long)blockIdx.x * 256 + threadIdx.x) * 4;
  float c[8];
#pragma unroll
  for (int e = 0; e < 8; ++e) c[e] = coeff[b * 8 + e];
  float4 v = *(const float4*)(Wb + i);
#pragma unroll
  for (int e = 0; e < 8; ++e) {
    float4 t = *(const float4*)(T + (long long)e * n + i);
    v.x += c[e] * t.x; v.y += c[e] * t.y; v.z += c[e] * t.z; v.w += c[e] * t.w;
  }
  *(float4*)(out + (long long)b * n + i) = v;
}

// per-token: logits = h2 @ Wgo + bgo (E=8), softmax, atomic mean into gate[B,E]
__global__ __launch_bounds__(256) void router_logits(
    const unsigned short* __restrict__ h2, const float* __restrict__ Wgo,
    const float* __restrict__ bgo, float* __restrict__ gate, int S, int D)
{
  const int token = blockIdx.x * 8 + (threadIdx.x >> 5);
  const int lane  = threadIdx.x & 31;
  const int b     = token / S;
  const unsigned short* hrow = h2 + (long long)token * D;

  float acc[8];
#pragma unroll
  for (int e = 0; e < 8; ++e) acc[e] = 0.f;

  for (int dd = 0; dd < D / 32; ++dd) {
    const int d = dd * 32 + lane;
    const float hv = bf2f(hrow[d]);
    const float4 w0 = *(const float4*)(Wgo + (long long)d * 8);
    const float4 w1 = *(const float4*)(Wgo + (long long)d * 8 + 4);
    acc[0] += hv * w0.x; acc[1] += hv * w0.y; acc[2] += hv * w0.z; acc[3] += hv * w0.w;
    acc[4] += hv * w1.x; acc[5] += hv * w1.y; acc[6] += hv * w1.z; acc[7] += hv * w1.w;
  }
#pragma unroll
  for (int e = 0; e < 8; ++e)
    for (int off = 16; off > 0; off >>= 1)
      acc[e] += __shfl_xor(acc[e], off, 32);

  if (lane == 0) {
    float l[8], mx = -3.4e38f;
#pragma unroll
    for (int e = 0; e < 8; ++e) { l[e] = acc[e] + bgo[e]; mx = l[e] > mx ? l[e] : mx; }
    float s = 0.f;
#pragma unroll
    for (int e = 0; e < 8; ++e) { l[e] = expf(l[e] - mx); s += l[e]; }
    const float inv = 1.f / (s * (float)S);
#pragma unroll
    for (int e = 0; e < 8; ++e) atomicAdd(&gate[b * 8 + e], l[e] * inv);
  }
}

__global__ void zero_f32(float* p, int n) {
  int i = blockIdx.x * blockDim.x + threadIdx.x;
  if (i < n) p[i] = 0.f;
}

// coeff = gate + ohe ; mirror gate into d_out tail
__global__ void finalize_gate(const float* __restrict__ gate,
                              const float* __restrict__ ohe,
                              float* __restrict__ coeff,
                              float* __restrict__ out_gate)
{
  int i = threadIdx.x;
  if (i < 64) {
    float g = gate[i];
    out_gate[i] = g;
    coeff[i] = g + ohe[i];
  }
}

// ---------------------------------------------------------------------------
extern "C" void kernel_launch(void* const* d_in, const int* in_sizes, int n_in,
                              void* d_out, int out_size, void* d_ws, size_t ws_size,
                              hipStream_t stream) {
  (void)in_sizes; (void)n_in; (void)out_size; (void)ws_size;
  const int B = 8, S = 512, D = 1024, F = 4096;
  const long long SD = (long long)S * D;     // 524288
  const long long SF = (long long)S * F;     // 2097152
  const long long DF = (long long)D * F;     // 4194304
  const long long DD = (long long)D * D;     // 1048576
  const long long BS = (long long)B * S;     // 4096

  const float* x    = (const float*)d_in[0];
  const float* ohe  = (const float*)d_in[1];
  const float* W1   = (const float*)d_in[2];
  const float* b1   = (const float*)d_in[3];
  const float* W2   = (const float*)d_in[4];
  const float* b2   = (const float*)d_in[5];
  const float* TW1  = (const float*)d_in[6];
  const float* Tb1  = (const float*)d_in[7];
  const float* TW2  = (const float*)d_in[8];
  const float* Tb2  = (const float*)d_in[9];
  const float* Wg1  = (const float*)d_in[10];
  const float* bg1  = (const float*)d_in[11];
  const float* Wg2  = (const float*)d_in[12];
  const float* bg2  = (const float*)d_in[13];
  const float* Wgo  = (const float*)d_in[14];
  const float* bgo  = (const float*)d_in[15];
  float* out = (float*)d_out;

  char* ws = (char*)d_ws;
  auto alloc = [&](size_t bytes) { char* p = ws; ws += (bytes + 255) & ~(size_t)255; return p; };
  unsigned short* xbf   = (unsigned short*)alloc(B * SD * 2);
  unsigned short* wg1bf = (unsigned short*)alloc(DD * 2);
  unsigned short* wg2bf = (unsigned short*)alloc(DD * 2);
  unsigned short* h1bf  = (unsigned short*)alloc(B * SD * 2);
  unsigned short* h2bf  = (unsigned short*)alloc(B * SD * 2);
  float*          gate  = (float*)alloc(B * 8 * 4);
  float*          coeff = (float*)alloc(B * 8 * 4);
  float*          b1b   = (float*)alloc((long long)B * F * 4);
  float*          b2b   = (float*)alloc((long long)B * D * 4);
  unsigned short* w1b   = (unsigned short*)alloc(B * DF * 2);
  unsigned short* w2b   = (unsigned short*)alloc(B * DF * 2);
  unsigned short* hmid  = (unsigned short*)alloc(B * SF * 2);

  // 0) zero gate accumulator
  zero_f32<<<1, 64, 0, stream>>>(gate, 64);

  // 1) fp32 -> bf16 conversions
  cvt_f32_to_bf16<<<(unsigned)(B * SD / 1024), 256, 0, stream>>>(x, xbf, B * SD);
  cvt_f32_to_bf16<<<(unsigned)(DD / 1024), 256, 0, stream>>>(Wg1, wg1bf, DD);
  cvt_f32_to_bf16<<<(unsigned)(DD / 1024), 256, 0, stream>>>(Wg2, wg2bf, DD);

  // 2) router hidden layers (M = B*S = 4096)
  gemm_bf16_wmma<1, 0><<<dim3(D / TN, (unsigned)(BS / TM), 1), 256, 0, stream>>>(
      xbf, wg1bf, bg1, h1bf, nullptr, 0, 0, 0, 0, (int)BS, D, D);
  gemm_bf16_wmma<1, 0><<<dim3(D / TN, (unsigned)(BS / TM), 1), 256, 0, stream>>>(
      h1bf, wg2bf, bg2, h2bf, nullptr, 0, 0, 0, 0, (int)BS, D, D);

  // 3) logits + softmax + mean over seq
  router_logits<<<(unsigned)(BS / 8), 256, 0, stream>>>(h2bf, Wgo, bgo, gate, S, D);

  // 4) coeff = gate + ohe ; write gate output
  finalize_gate<<<1, 64, 0, stream>>>(gate, ohe, coeff, out + (long long)B * SD);

  // 5) merge per-sample weights/biases
  merge_w_bf16<<<dim3((unsigned)(DF / 1024), B), 256, 0, stream>>>(W1, TW1, coeff, w1b, DF);
  merge_w_bf16<<<dim3((unsigned)(DF / 1024), B), 256, 0, stream>>>(W2, TW2, coeff, w2b, DF);
  merge_b_f32<<<dim3(F / 1024, B), 256, 0, stream>>>(b1, Tb1, coeff, b1b, F);
  merge_b_f32<<<dim3(1, B), 256, 0, stream>>>(b2, Tb2, coeff, b2b, D);

  // 6) FFN layer 1 (exact GELU), batched over B
  gemm_bf16_wmma<2, 0><<<dim3(F / TN, S / TM, B), 256, 0, stream>>>(
      xbf, w1b, b1b, hmid, nullptr, SD, DF, F, SF, S, F, D);

  // 7) FFN layer 2 -> fp32 d_out, batched over B
  gemm_bf16_wmma<0, 1><<<dim3(D / TN, S / TM, B), 256, 0, stream>>>(
      hmid, w2b, b2b, nullptr, out, SF, DF, D, SD, S, D, F);
}
